// StatefulCausalDiffHopfieldAttentionHead_34342558498822
// MI455X (gfx1250) — matrix operand order repaired and checked by reading, not compiled, not run on hardware
//
#include <hip/hip_runtime.h>
#include <hip/hip_bf16.h>

#define T_TOT 2304
#define BATCH 2
#define DIN   1024
#define DH    128
#define D2    256
#define SLEN  128

static constexpr float SCALE_QK    = 0.08838834764831845f;   // 1/sqrt(128)
static constexpr float LAMBDA_INIT = 0.6192834728526787f;    // 0.8 - 0.6*exp(-1.2)

// flash kernel LDS layout
#define LI_BYTES   (16 * T_TOT * 4)        // 147456 (COPE only)
#define PSTG_BYTES (4 * 16 * 32 * 2)       // 4096
#define KBUF_HALFS (32 * 128)              // one K tile: 32 rows x 128 halfs
#define KBUF_BYTES (2 * KBUF_HALFS * 2)    // 16384 (double buffered)

typedef __attribute__((ext_vector_type(16))) _Float16 v16h;
typedef __attribute__((ext_vector_type(8)))  float    v8f;

// ---------------- WMMA fragment loaders (CDNA5 wave32 layouts) ----------------
__device__ __forceinline__ v16h load_a16(const _Float16* base, int lda, int k0, int lane) {
  const int row = lane & 15;
  const _Float16* p = base + (long long)row * lda + k0 + ((lane >> 4) << 3);
  v16h a;
#pragma unroll
  for (int e = 0; e < 8; ++e) { a[e] = p[e]; a[e + 8] = p[e + 16]; }
  return a;
}
__device__ __forceinline__ v16h load_b16(const _Float16* base, int ldb, int k0, int lane) {
  const int col = lane & 15;
  const _Float16* p = base + (long long)col * ldb + k0 + ((lane >> 4) << 3);
  v16h b;
#pragma unroll
  for (int e = 0; e < 8; ++e) { b[e] = p[e]; b[e + 8] = p[e + 16]; }
  return b;
}
__device__ __forceinline__ v8f wmma16(v16h a, v16h b, v8f c) {
  return __builtin_amdgcn_wmma_f32_16x16x32_f16(false, a, false, b, (short)0, c, false, false);
}

// ---------------- gfx1250 async copy helpers ----------------
// Stage one 32x128-half K tile (8 KB) into LDS with 128 threads x 4 b128 async loads.
__device__ __forceinline__ void stage_k_async(const _Float16* kh, int s0, _Float16* buf, int tid) {
  const uint32_t ldsBase = (uint32_t)(uintptr_t)buf;  // low 32 bits of flat ptr == LDS byte addr
#pragma unroll
  for (int i = 0; i < 4; ++i) {
    const int c = tid * 4 + i;        // 16-byte chunk index, 512 total
    const int row = c >> 4;           // 16 chunks per 256B row
    const int col = c & 15;
    const unsigned long long g =
        (unsigned long long)(uintptr_t)((const char*)(kh + (long long)(s0 + row) * D2) + col * 16);
    const uint32_t lds = ldsBase + row * 256 + col * 16;
    asm volatile("global_load_async_to_lds_b128 %0, %1, off"
                 :: "v"(lds), "v"(g) : "memory");
  }
}
__device__ __forceinline__ void wait_async0() {
  asm volatile("s_wait_asynccnt 0x0" ::: "memory");
}

// ---------------- generic WMMA GEMM: C = A(M,K) @ Bt(N,K)^T (+bias) ----------------
// one wave per 16x16 tile; grid = (N/16, M/16, batch)
// outMode: 0 = f32 row-major, 1 = f16 row-major, 2 = f16 transposed (out[col*ldc+row])
__global__ void gemm_wmma_kernel(const _Float16* __restrict__ A, int lda, long long aStrideB,
                                 const _Float16* __restrict__ Bt, int ldb,
                                 void* __restrict__ Cout, int ldc, long long cStrideB,
                                 int K, const float* __restrict__ bias, int outMode) {
  const int lane = threadIdx.x & 31;
  const int nt = blockIdx.x, mt = blockIdx.y, bz = blockIdx.z;
  const _Float16* Ab = A + (long long)bz * aStrideB + (long long)mt * 16 * lda;
  const _Float16* Bb = Bt + (long long)nt * 16 * ldb;
  v8f acc = {};
  for (int k0 = 0; k0 < K; k0 += 32) {
    v16h a = load_a16(Ab, lda, k0, lane);
    v16h b = load_b16(Bb, ldb, k0, lane);
    acc = wmma16(a, b, acc);
  }
  const int half = lane >> 4, col = lane & 15;
  const int gcol = nt * 16 + col;
  const float bs = bias ? bias[gcol] : 0.0f;
#pragma unroll
  for (int r = 0; r < 8; ++r) {
    const int grow = mt * 16 + r + 8 * half;
    const float v = acc[r] + bs;
    if (outMode == 0)
      ((float*)Cout)[(long long)bz * cStrideB + (long long)grow * ldc + gcol] = v;
    else if (outMode == 1)
      ((_Float16*)Cout)[(long long)bz * cStrideB + (long long)grow * ldc + gcol] = (_Float16)v;
    else
      ((_Float16*)Cout)[(long long)bz * cStrideB + (long long)gcol * ldc + grow] = (_Float16)v;
  }
}

// ---------------- flash attention (descending s) with optional CoPE bias ----------------
// grid = (T/16, B), block = 128 (4 waves, each owns 64 V-columns).
// K tiles are staged once per block into LDS via async copies (double buffered);
// all 4 waves consume the shared tile for the QK^T WMMAs.
template <bool COPE>
__global__ void flash_kernel(const _Float16* __restrict__ qh,
                             const _Float16* __restrict__ kh,
                             const _Float16* __restrict__ vT,
                             const _Float16* __restrict__ embT,
                             float* __restrict__ outAtt) {
  extern __shared__ char smem[];
  float* li = (float*)smem;                                       // 16 x T (COPE only)
  _Float16* pstgAll = (_Float16*)(smem + (COPE ? LI_BYTES : 0));
  _Float16* kbuf = (_Float16*)(smem + (COPE ? LI_BYTES : 0) + PSTG_BYTES);
  const int tid = threadIdx.x;
  const int lane = tid & 31;
  const int wave = tid >> 5;
  _Float16* pstg = pstgAll + wave * 16 * 32;
  const int b = blockIdx.y;
  const int trow0 = blockIdx.x * 16;
  qh += (long long)b * T_TOT * D2;
  kh += (long long)b * T_TOT * D2;
  vT += (long long)b * D2 * T_TOT;
  outAtt += (long long)b * T_TOT * D2;
  const int half = lane >> 4, lrow = lane & 15;
  const int rowoff = 8 * half;
  (void)embT; (void)li;

  v16h qa[4];
#pragma unroll
  for (int kt = 0; kt < 4; ++kt) qa[kt] = load_a16(qh + (long long)trow0 * D2, D2, kt * 32, lane);

  if constexpr (COPE) {
    // li[t, s2] = q[t] . emb[:, s2] for this 16-row block, via WMMA into LDS
    for (int s2t = wave; s2t < T_TOT / 16; s2t += 4) {
      v8f acc = {};
#pragma unroll
      for (int kt = 0; kt < 4; ++kt) {
        v16h bb = load_b16(embT + (long long)s2t * 16 * DH, DH, kt * 32, lane);
        acc = wmma16(qa[kt], bb, acc);
      }
#pragma unroll
      for (int r = 0; r < 8; ++r)
        li[(r + rowoff) * T_TOT + s2t * 16 + lrow] = acc[r];
    }
    __syncthreads();
  }

  float m[8], l[8], carry[8];
  v8f zeroV = {};
  v8f o[4];
#pragma unroll
  for (int r = 0; r < 8; ++r) { m[r] = -1e30f; l[r] = 0.0f; carry[r] = 0.0f; }
#pragma unroll
  for (int n = 0; n < 4; ++n) o[n] = zeroV;

  const int tmax = trow0 + 15;
  const int stLo = COPE ? (T_TOT / 32 - 1) : (tmax >> 5);

  // prime the async K-tile pipeline
  stage_k_async(kh, stLo * 32, kbuf, tid);
  int curBuf = 0;

  for (int st = stLo; st >= 0; --st) {
    const int s0 = st * 32;
    wait_async0();
    __syncthreads();                 // whole K tile visible to all waves
    if (st > 0) stage_k_async(kh, (st - 1) * 32, kbuf + (curBuf ^ 1) * KBUF_HALFS, tid);
    const _Float16* kb = kbuf + curBuf * KBUF_HALFS;

    v8f sr[2];
#pragma unroll
    for (int sub = 0; sub < 2; ++sub) {
      v8f acc = {};
#pragma unroll
      for (int kt = 0; kt < 4; ++kt) {
        v16h bb = load_b16(kb + sub * 16 * 128, 128, kt * 32, lane);
        acc = wmma16(qa[kt], bb, acc);
      }
      sr[sub] = acc;
    }

    float cbias[2][8];
    if constexpr (COPE) {
      // descending suffix-sum of sigmoid gates; high sub-tile first
#pragma unroll
      for (int subi = 0; subi < 2; ++subi) {
        const int sub = 1 - subi;
#pragma unroll
        for (int r = 0; r < 8; ++r) {
          float g = 1.0f / (1.0f + __expf(-sr[sub][r]));
          float v = g;
#pragma unroll
          for (int d = 1; d < 16; d <<= 1) {
            float t = __shfl_down(v, d, 16);
            if (lrow + d < 16) v += t;
          }
          float total = __shfl(v, 0, 16);
          float pos = v + carry[r];
          carry[r] += total;
          pos = fminf(pos, (float)(T_TOT - 1));
          float pf = floorf(pos);
          int ipf = (int)pf;
          int ipc = ipf + 1; if (ipc > T_TOT - 1) ipc = T_TOT - 1;
          const int rowbase = (r + rowoff) * T_TOT;
          float lf = li[rowbase + ipf];
          float lc = li[rowbase + ipc];
          float w = pos - pf;
          cbias[sub][r] = lc * w + lf * (1.0f - w);
        }
      }
    }

    if (s0 <= tmax) {  // tile contributes to softmax (block-uniform condition)
      float logit[2][8], tmaxr[8];
#pragma unroll
      for (int r = 0; r < 8; ++r) tmaxr[r] = -1e30f;
#pragma unroll
      for (int sub = 0; sub < 2; ++sub) {
#pragma unroll
        for (int r = 0; r < 8; ++r) {
          const int s = s0 + sub * 16 + lrow;
          const int t = trow0 + r + rowoff;
          float bv = 0.0f;
          if constexpr (COPE) bv = cbias[sub][r];
          float lv = (s <= t) ? (sr[sub][r] * SCALE_QK + bv) : -1e30f;
          logit[sub][r] = lv;
          float mv = lv;
#pragma unroll
          for (int d = 1; d < 16; d <<= 1) mv = fmaxf(mv, __shfl_xor(mv, d, 16));
          tmaxr[r] = fmaxf(tmaxr[r], mv);
        }
      }
      float scl[8], rs[8];
#pragma unroll
      for (int r = 0; r < 8; ++r) {
        float mn = fmaxf(m[r], tmaxr[r]);
        scl[r] = __expf(m[r] - mn);
        m[r] = mn;
        rs[r] = 0.0f;
      }
#pragma unroll
      for (int sub = 0; sub < 2; ++sub) {
#pragma unroll
        for (int r = 0; r < 8; ++r) {
          float p = (logit[sub][r] > -1e29f) ? __expf(logit[sub][r] - m[r]) : 0.0f;
          float ps = p;
#pragma unroll
          for (int d = 1; d < 16; d <<= 1) ps += __shfl_xor(ps, d, 16);
          rs[r] += ps;
          pstg[(r + rowoff) * 32 + sub * 16 + lrow] = (_Float16)p;
        }
      }
#pragma unroll
      for (int r = 0; r < 8; ++r) l[r] = l[r] * scl[r] + rs[r];
#pragma unroll
      for (int n = 0; n < 4; ++n)
#pragma unroll
        for (int r = 0; r < 8; ++r) o[n][r] *= scl[r];

      v16h pa = load_a16(pstg, 32, 0, lane);  // P tile (16x32) in A layout
#pragma unroll
      for (int n = 0; n < 4; ++n) {
        v16h bb = load_b16(vT + (long long)(wave * 64 + n * 16) * T_TOT, T_TOT, s0, lane);
        o[n] = wmma16(pa, bb, o[n]);
      }
    }
    __syncthreads();   // protect K buffer reuse two iterations ahead
    curBuf ^= 1;
  }

#pragma unroll
  for (int r = 0; r < 8; ++r) {
    const float inv = (l[r] > 0.0f) ? 1.0f / l[r] : 0.0f;
    const int t = trow0 + r + rowoff;
#pragma unroll
    for (int n = 0; n < 4; ++n)
      outAtt[(long long)t * D2 + wave * 64 + n * 16 + lrow] = o[n][r] * inv;
  }
}

// ---------------- small helper kernels ----------------
__global__ void ln_in_kernel(const float* __restrict__ x,
                             const float* wss, const float* bss,
                             const float* wm,  const float* bm,
                             const float* wse, const float* bse,
                             _Float16* __restrict__ xn) {
  const int row = blockIdx.x;                 // 0..B*T-1
  const int t = row % T_TOT;
  const float* w; const float* b;
  if (t < SLEN)              { w = wss; b = bss; }
  else if (t < T_TOT - SLEN) { w = wm;  b = bm;  }
  else                       { w = wse; b = bse; }
  const float* xr = x + (long long)row * DIN;
  const int tid = threadIdx.x;
  float vals[4]; float s = 0.0f;
#pragma unroll
  for (int i = 0; i < 4; ++i) { vals[i] = xr[i * 256 + tid]; s += vals[i]; }
  __shared__ float red[256];
  red[tid] = s; __syncthreads();
  for (int d = 128; d > 0; d >>= 1) { if (tid < d) red[tid] += red[tid + d]; __syncthreads(); }
  const float mean = red[0] * (1.0f / DIN);
  __syncthreads();
  float s2 = 0.0f;
#pragma unroll
  for (int i = 0; i < 4; ++i) { float dd = vals[i] - mean; s2 += dd * dd; }
  red[tid] = s2; __syncthreads();
  for (int d = 128; d > 0; d >>= 1) { if (tid < d) red[tid] += red[tid + d]; __syncthreads(); }
  const float rstd = rsqrtf(red[0] * (1.0f / DIN) + 1e-5f);
#pragma unroll
  for (int i = 0; i < 4; ++i) {
    const int j = i * 256 + tid;
    xn[(long long)row * DIN + j] = (_Float16)((vals[i] - mean) * rstd * w[j] + b[j]);
  }
}

__global__ void ln_out_kernel(const float* __restrict__ in, const float* w, const float* b,
                              float* __restrict__ out) {
  const int row = blockIdx.x;
  const int tid = threadIdx.x;
  const float v = in[(long long)row * D2 + tid];
  __shared__ float red[256];
  red[tid] = v; __syncthreads();
  for (int d = 128; d > 0; d >>= 1) { if (tid < d) red[tid] += red[tid + d]; __syncthreads(); }
  const float mean = red[0] * (1.0f / D2);
  __syncthreads();
  const float dd = v - mean;
  red[tid] = dd * dd; __syncthreads();
  for (int d = 128; d > 0; d >>= 1) { if (tid < d) red[tid] += red[tid + d]; __syncthreads(); }
  const float rstd = rsqrtf(red[0] * (1.0f / D2) + 1e-5f);
  out[(long long)row * D2 + tid] = dd * rstd * w[tid] + b[tid];
}

__global__ void cast_transpose_kernel(const float* __restrict__ in, _Float16* __restrict__ out,
                                      int K, int N) {
  const int i = blockIdx.x * blockDim.x + threadIdx.x;
  if (i >= K * N) return;
  const int k = i / N, n = i % N;
  out[(long long)n * K + k] = (_Float16)in[i];
}

__global__ void lambda_kernel(const float* q1, const float* k1, const float* q2, const float* k2,
                              float* lam) {
  __shared__ float r1[128], r2[128];
  const int t = threadIdx.x;
  r1[t] = q1[t] * k1[t]; r2[t] = q2[t] * k2[t];
  __syncthreads();
  for (int d = 64; d > 0; d >>= 1) { if (t < d) { r1[t] += r1[t + d]; r2[t] += r2[t + d]; } __syncthreads(); }
  if (t == 0) lam[0] = __expf(r1[0]) - __expf(r2[0]) + LAMBDA_INIT;
}

__global__ void combine_kernel(const float* __restrict__ a1, const float* __restrict__ a2,
                               const float* __restrict__ lam, float* __restrict__ mf,
                               _Float16* __restrict__ mf16, int n) {
  const int i = blockIdx.x * blockDim.x + threadIdx.x;
  if (i >= n) return;
  const float v = a1[i] - lam[0] * a2[i];
  mf[i] = v; mf16[i] = (_Float16)v;
}

// ---------------- host ----------------
extern "C" void kernel_launch(void* const* d_in, const int* in_sizes, int n_in,
                              void* d_out, int out_size, void* d_ws, size_t ws_size,
                              hipStream_t stream) {
  (void)in_sizes; (void)n_in; (void)out_size; (void)ws_size;
  const float* x = (const float*)d_in[0];

  char* ws = (char*)d_ws;
  size_t off = 0;
  auto alloc = [&](size_t bytes) -> char* {
    char* p = ws + off;
    off = (off + bytes + 255) & ~(size_t)255;
    return p;
  };
  _Float16* xn16  = (_Float16*)alloc((size_t)BATCH * T_TOT * DIN * 2);
  _Float16* wh16  = (_Float16*)alloc((size_t)3 * DH * DIN * 2);
  _Float16* wq16  = (_Float16*)alloc((size_t)3 * D2 * DH * 2);
  _Float16* wk16  = (_Float16*)alloc((size_t)3 * D2 * DH * 2);
  _Float16* wv16  = (_Float16*)alloc((size_t)3 * D2 * DH * 2);
  _Float16* wd16  = (_Float16*)alloc((size_t)3 * DH * D2 * 2);
  _Float16* emb16 = (_Float16*)alloc((size_t)2 * T_TOT * DH * 2);
  _Float16* mem16 = (_Float16*)alloc((size_t)BATCH * T_TOT * DH * 2);
  _Float16* q16   = (_Float16*)alloc((size_t)BATCH * T_TOT * D2 * 2);
  _Float16* k16   = (_Float16*)alloc((size_t)BATCH * T_TOT * D2 * 2);
  _Float16* vT16  = (_Float16*)alloc((size_t)BATCH * D2 * T_TOT * 2);
  float*    att1  = (float*)alloc((size_t)BATCH * T_TOT * D2 * 4);
  float*    att2  = (float*)alloc((size_t)BATCH * T_TOT * D2 * 4);
  float*    mf32  = (float*)alloc((size_t)BATCH * T_TOT * D2 * 4);
  _Float16* mf16  = (_Float16*)alloc((size_t)BATCH * T_TOT * D2 * 2);
  float*    lam   = (float*)alloc(256);

  // 1) input LN -> f16
  ln_in_kernel<<<BATCH * T_TOT, 256, 0, stream>>>(
      x, (const float*)d_in[3], (const float*)d_in[4], (const float*)d_in[1],
      (const float*)d_in[2], (const float*)d_in[5], (const float*)d_in[6], xn16);

  // 2) weight casts/transposes to (N,K) f16
  auto castT = [&](const float* in, _Float16* out, int K, int N) {
    const int tot = K * N;
    cast_transpose_kernel<<<(tot + 255) / 256, 256, 0, stream>>>(in, out, K, N);
  };
  const int hid_i[3] = {8, 7, 9};
  const int q_i[3] = {14, 11, 17}, k_i[3] = {13, 10, 16}, v_i[3] = {15, 12, 18};
  const int d_i[3] = {21, 19, 23};
  for (int s = 0; s < 3; ++s) {
    castT((const float*)d_in[hid_i[s]], wh16 + (size_t)s * DH * DIN, DIN, DH);
    castT((const float*)d_in[q_i[s]],   wq16 + (size_t)s * D2 * DH, DH, D2);
    castT((const float*)d_in[k_i[s]],   wk16 + (size_t)s * D2 * DH, DH, D2);
    castT((const float*)d_in[v_i[s]],   wv16 + (size_t)s * D2 * DH, DH, D2);
    castT((const float*)d_in[d_i[s]],   wd16 + (size_t)s * DH * D2, D2, DH);
  }
  castT((const float*)d_in[29], emb16, DH, T_TOT);
  castT((const float*)d_in[30], emb16 + (size_t)T_TOT * DH, DH, T_TOT);

  // 3) lambda scalar
  lambda_kernel<<<1, 128, 0, stream>>>((const float*)d_in[25], (const float*)d_in[27],
                                       (const float*)d_in[26], (const float*)d_in[28], lam);

  const int segM[3]   = {SLEN, T_TOT - 2 * SLEN, SLEN};
  const int segOff[3] = {0, SLEN, T_TOT - SLEN};

  // 4) hidden projections: xn16 @ Wh -> mem16 (f16)
  for (int s = 0; s < 3; ++s) {
    dim3 g(DH / 16, segM[s] / 16, BATCH);
    gemm_wmma_kernel<<<g, 32, 0, stream>>>(
        xn16 + (size_t)segOff[s] * DIN, DIN, (long long)T_TOT * DIN,
        wh16 + (size_t)s * DH * DIN, DIN,
        mem16 + (size_t)segOff[s] * DH, DH, (long long)T_TOT * DH,
        DIN, nullptr, 1);
  }

  const float* dbias[3] = {(const float*)d_in[22], (const float*)d_in[20], (const float*)d_in[24]};
  const size_t copeSm  = (size_t)LI_BYTES + PSTG_BYTES + KBUF_BYTES;
  const size_t plainSm = (size_t)PSTG_BYTES + KBUF_BYTES;
  (void)hipFuncSetAttribute(reinterpret_cast<const void*>(&flash_kernel<true>),
                            hipFuncAttributeMaxDynamicSharedMemorySize, (int)copeSm);
  (void)hipFuncSetAttribute(reinterpret_cast<const void*>(&flash_kernel<false>),
                            hipFuncAttributeMaxDynamicSharedMemorySize, (int)plainSm);

  for (int ix = 0; ix < 2; ++ix) {
    // q/k/v projections; v written transposed (256,T)
    for (int s = 0; s < 3; ++s) {
      dim3 g(D2 / 16, segM[s] / 16, BATCH);
      gemm_wmma_kernel<<<g, 32, 0, stream>>>(
          mem16 + (size_t)segOff[s] * DH, DH, (long long)T_TOT * DH,
          wq16 + (size_t)s * D2 * DH, DH,
          q16 + (size_t)segOff[s] * D2, D2, (long long)T_TOT * D2, DH, nullptr, 1);
      gemm_wmma_kernel<<<g, 32, 0, stream>>>(
          mem16 + (size_t)segOff[s] * DH, DH, (long long)T_TOT * DH,
          wk16 + (size_t)s * D2 * DH, DH,
          k16 + (size_t)segOff[s] * D2, D2, (long long)T_TOT * D2, DH, nullptr, 1);
      gemm_wmma_kernel<<<g, 32, 0, stream>>>(
          mem16 + (size_t)segOff[s] * DH, DH, (long long)T_TOT * DH,
          wv16 + (size_t)s * D2 * DH, DH,
          vT16 + (size_t)segOff[s], T_TOT, (long long)D2 * T_TOT, DH, nullptr, 2);
    }
    // attention heads
    dim3 ga(T_TOT / 16, BATCH);
    if (ix == 0) {
      flash_kernel<true><<<ga, 128, copeSm, stream>>>(q16, k16, vT16, emb16, att1);
      flash_kernel<true><<<ga, 128, copeSm, stream>>>(q16 + DH, k16 + DH, vT16,
                                                      emb16 + (size_t)T_TOT * DH, att2);
    } else {
      flash_kernel<false><<<ga, 128, plainSm, stream>>>(q16, k16, vT16, nullptr, att1);
      flash_kernel<false><<<ga, 128, plainSm, stream>>>(q16 + DH, k16 + DH, vT16, nullptr, att2);
    }
    // diff combine
    const int ncomb = BATCH * T_TOT * D2;
    combine_kernel<<<(ncomb + 255) / 256, 256, 0, stream>>>(att1, att2, lam, mf32, mf16, ncomb);
    // down-projection after first iteration
    if (ix == 0) {
      for (int s = 0; s < 3; ++s) {
        dim3 g(DH / 16, segM[s] / 16, BATCH);
        gemm_wmma_kernel<<<g, 32, 0, stream>>>(
            mf16 + (size_t)segOff[s] * D2, D2, (long long)T_TOT * D2,
            wd16 + (size_t)s * DH * D2, D2,
            mem16 + (size_t)segOff[s] * DH, DH, (long long)T_TOT * DH,
            D2, dbias[s], 1);
      }
    }
  }

  // 5) output LN
  ln_out_kernel<<<BATCH * T_TOT, 256, 0, stream>>>(mf32, (const float*)d_in[31],
                                                   (const float*)d_in[32], (float*)d_out);
}
